// MLA_10599979286547
// MI455X (gfx1250) — matrix operand order repaired and checked by reading, not compiled, not run on hardware
//
#include <hip/hip_runtime.h>

// ---------------- types / helpers ----------------
typedef unsigned short u16;
typedef unsigned int u32;
typedef __attribute__((ext_vector_type(16))) __bf16 v16bf;
typedef __attribute__((ext_vector_type(8))) float v8f;
typedef __attribute__((ext_vector_type(4))) unsigned int u32x4;
typedef __attribute__((ext_vector_type(8))) int i32x8;
typedef __attribute__((ext_vector_type(4))) int i32x4;

#if __has_builtin(__builtin_amdgcn_tensor_load_to_lds) && \
    __has_builtin(__builtin_amdgcn_s_wait_tensorcnt)
#define USE_TDM 1
#endif

struct Pack32 { uint4 lo, hi; };

static __device__ __forceinline__ v16bf frag_from(uint4 lo, uint4 hi) {
  Pack32 p{lo, hi};
  return __builtin_bit_cast(v16bf, p);
}

static __device__ __forceinline__ u16 f2bf(float f) {
  u32 u = __float_as_uint(f);
  u32 r = u + 0x7fffu + ((u >> 16) & 1u);   // round to nearest even
  return (u16)(r >> 16);
}
static __device__ __forceinline__ float bf2f(u16 h) {
  return __uint_as_float(((u32)h) << 16);
}

// ---------------- problem constants ----------------
constexpr int Bc = 4, Sc = 2048, Dc = 2048;
constexpr int Hc = 16, NOPEc = 128, ROPEc = 64, Vc = 128, KVRc = 512;
constexpr int QKc = NOPEc + ROPEc;          // 192
constexpr int Mc = Bc * Sc;                 // 8192 tokens

#ifdef USE_TDM
// Issue one TDM load of a 64x32 bf16 tile (row stride = lda elements) into LDS.
// LDS destination gets 16B of padding every 64B row => 80B row stride, matching
// the padded fragment layout used by the compute waves.
// D# fields per CDNA5 ISA ch.8 (group0: count/lds/global/type; group1: sizes).
static __device__ __forceinline__ void tdm_load_tile(u32 lds_off, const u16* gp,
                                                     u32 lda_elems) {
  unsigned long long ga = (unsigned long long)gp;
  u32x4 g0;
  g0.x = 1u;                                             // count=1 (valid), user mode
  g0.y = lds_off;                                        // lds_addr (bytes)
  g0.z = (u32)ga;                                        // global_addr[31:0]
  g0.w = (u32)((ga >> 32) & 0x01FFFFFFu) | (2u << 30);   // global_addr[56:32], type=2

  u32 w0 = (1u << 16)    // data_size = 2 bytes
         | (1u << 20)    // pad_enable
         | (3u << 22)    // pad_interval: 16 DWORDs (64B) between pads
         | (3u << 25);   // pad_amount: 4 DWORDs (16B)
  u32 w1 = (32u & 0xffffu) << 16;                 // tensor_dim0[15:0] = 32
  u32 w2 = ((64u & 0xffffu) << 16);               // tensor_dim0 hi=0 | tensor_dim1 lo=64
  u32 w3 = (32u << 16);                           // tensor_dim1 hi=0 | tile_dim0=32
  u32 w4 = 64u;                                   // tile_dim1=64, tile_dim2=0
  u32 w5 = lda_elems;                             // tensor_dim0_stride[31:0]
  u32 w6 = 0u;                                    // stride hi, dim1_stride lo
  u32 w7 = 0u;
  i32x8 g1 = { (int)w0, (int)w1, (int)w2, (int)w3,
               (int)w4, (int)w5, (int)w6, (int)w7 };
  i32x4 gz = { 0, 0, 0, 0 };
#if __clang_major__ >= 23
  i32x8 gz8 = { 0, 0, 0, 0, 0, 0, 0, 0 };
  __builtin_amdgcn_tensor_load_to_lds(g0, g1, gz, gz, gz8, 0);
#else
  __builtin_amdgcn_tensor_load_to_lds(g0, g1, gz, gz, 0);
#endif
}
static __device__ __forceinline__ u32 lds_off_of(const void* p) {
  // generic pointer to LDS: low 32 bits are the wave-relative LDS byte offset
  return (u32)(unsigned long long)p;
}
#endif

// ---------------- fp32 -> bf16 converter (4 elems/thread) ----------------
__global__ __launch_bounds__(256)
void k_cvt(const float* __restrict__ in, u16* __restrict__ out, int n) {
  int i = (blockIdx.x * 256 + threadIdx.x) * 4;
  if (i >= n) return;
  float4 v = *(const float4*)(in + i);
  u32 a = (u32)f2bf(v.x) | ((u32)f2bf(v.y) << 16);
  u32 b = (u32)f2bf(v.z) | ((u32)f2bf(v.w) << 16);
  uint2 r; r.x = a; r.y = b;
  *(uint2*)(out + i) = r;
}

// wkv_b (H,256,512) fp32 -> nopeT (H,512,128) bf16 (transposed) + vpart (H,128,512) bf16
__global__ __launch_bounds__(256)
void k_cvt_wkvb(const float* __restrict__ w, u16* __restrict__ nopeT,
                u16* __restrict__ vpart) {
  int idx = blockIdx.x * 256 + threadIdx.x;     // H*256*512
  if (idx >= Hc * 256 * KVRc) return;
  int c = idx & (KVRc - 1);
  int r = (idx >> 9) & 255;
  int h = idx >> 17;
  u16 b = f2bf(w[idx]);
  if (r < NOPEc) nopeT[h * (KVRc * NOPEc) + c * NOPEc + r] = b;
  else           vpart[h * (Vc * KVRc) + (r - NOPEc) * KVRc + c] = b;
}

// ---------------- generic NT GEMM: C[M,N] = A[M,K] * B[N,K]^T (bf16 in, f32 acc) ----
// block = 128 threads = 4 waves; tile 64x64, BK=32; batch via gridDim.z with strides.
// Tiles staged by the Tensor Data Mover (double-buffered, TENSORcnt-tracked) when
// available; manual global->LDS staging otherwise.
template <bool OUT_BF16>
__global__ __launch_bounds__(128)
void k_gemm_nt(const u16* __restrict__ A, const u16* __restrict__ Bm,
               void* __restrict__ Cv, int K, int lda, int ldb, int ldc,
               long long sA, long long sB, long long sC) {
  __shared__ u16 As[2][64 * 40];   // +8 bf16 pad per row: 80B stride, conflict-free
  __shared__ u16 Bs[2][64 * 40];
  const u16* Ab = A + (long long)blockIdx.z * sA + (size_t)blockIdx.y * 64 * lda;
  const u16* Bb = Bm + (long long)blockIdx.z * sB + (size_t)blockIdx.x * 64 * ldb;
  const int tid = threadIdx.x;
  const int lane = tid & 31, w = tid >> 5;
  const int lg = lane >> 4, lr = lane & 15;

  v8f acc[4] = {};
  const int ntiles = K >> 5;

#ifdef USE_TDM
  if (w == 0) {
    tdm_load_tile(lds_off_of(&As[0][0]), Ab, (u32)lda);
    tdm_load_tile(lds_off_of(&Bs[0][0]), Bb, (u32)ldb);
  }
#endif

  for (int i = 0; i < ntiles; ++i) {
    const int cur = i & 1;
    const int nxt = cur ^ 1;
    const int k0 = i * 32;

#ifdef USE_TDM
    if (w == 0) {
      if (i + 1 < ntiles) {
        tdm_load_tile(lds_off_of(&As[nxt][0]), Ab + (k0 + 32), (u32)lda);
        tdm_load_tile(lds_off_of(&Bs[nxt][0]), Bb + (k0 + 32), (u32)ldb);
        __builtin_amdgcn_s_wait_tensorcnt(2);   // current tile's 2 loads done
      } else {
        __builtin_amdgcn_s_wait_tensorcnt(0);
      }
    }
#else
#pragma unroll
    for (int p = 0; p < 2; ++p) {
      int row = (tid >> 2) + p * 32;
      int ch = (tid & 3) * 8;
      *(uint4*)&As[cur][row * 40 + ch] = *(const uint4*)(Ab + (size_t)row * lda + k0 + ch);
      *(uint4*)&Bs[cur][row * 40 + ch] = *(const uint4*)(Bb + (size_t)row * ldb + k0 + ch);
    }
#endif
    __syncthreads();

    const u16* Asc = As[cur];
    const u16* Bsc = Bs[cur];
    const int ra = w * 16 + lr;  // A-frag: lane row, K chunks [lg*8,+8) & [16+lg*8,+8)
    v16bf af = frag_from(*(const uint4*)&Asc[ra * 40 + lg * 8],
                         *(const uint4*)&Asc[ra * 40 + 16 + lg * 8]);
#pragma unroll
    for (int nb = 0; nb < 4; ++nb) {
      int rb = nb * 16 + lr;     // B-frag: lane col, K chunks [lg*8,+8) & [16+lg*8,+8)
      v16bf bf = frag_from(*(const uint4*)&Bsc[rb * 40 + lg * 8],
                           *(const uint4*)&Bsc[rb * 40 + 16 + lg * 8]);
      acc[nb] = __builtin_amdgcn_wmma_f32_16x16x32_bf16(false, af, false, bf,
                                                        (short)0, acc[nb], false, false);
    }
    __syncthreads();
  }

  const size_t crow0 = (size_t)blockIdx.y * 64 + w * 16 + lg * 8;
  const int ccol0 = blockIdx.x * 64 + lr;
#pragma unroll
  for (int nb = 0; nb < 4; ++nb) {
    const float* f = (const float*)&acc[nb];
#pragma unroll
    for (int r = 0; r < 8; ++r) {
      size_t idx = (crow0 + r) * (size_t)ldc + ccol0 + nb * 16;
      if (OUT_BF16) ((u16*)Cv)[(long long)blockIdx.z * sC + idx] = f2bf(f[r]);
      else          ((float*)Cv)[(long long)blockIdx.z * sC + idx] = f[r];
    }
  }
}

// ---------------- kv post: RMSNorm(512) -> kv_c bf16 ; RoPE(64) -> k_pe bf16 -------
__global__ __launch_bounds__(256)
void k_kvpost(const float* __restrict__ kvf, const float* __restrict__ nw,
              const float* __restrict__ fcos, const float* __restrict__ fsin,
              u16* __restrict__ kvc, u16* __restrict__ kpe) {
  const int lane = threadIdx.x & 31;
  const int w = threadIdx.x >> 5;
  const int m = blockIdx.x * 8 + w;          // one token per wave
  const float* row = kvf + (size_t)m * (KVRc + ROPEc);
  float vals[16];
  float ss = 0.f;
#pragma unroll
  for (int j = 0; j < 16; ++j) {
    float v = row[j * 32 + lane];
    vals[j] = v;
    ss += v * v;
  }
#pragma unroll
  for (int o = 1; o <= 16; o <<= 1) ss += __shfl_xor(ss, o, 32);
  float scl = rsqrtf(ss * (1.f / (float)KVRc) + 1e-6f);
#pragma unroll
  for (int j = 0; j < 16; ++j) {
    int c = j * 32 + lane;
    kvc[(size_t)m * KVRc + c] = f2bf(vals[j] * scl * nw[c]);
  }
  // RoPE on the 64-dim pe tail; 32 pairs -> one pair per lane
  int s = m & (Sc - 1);
  float x1 = row[KVRc + 2 * lane];
  float x2 = row[KVRc + 2 * lane + 1];
  float co = fcos[s * 32 + lane], si = fsin[s * 32 + lane];
  kpe[(size_t)m * ROPEc + 2 * lane]     = f2bf(x1 * co - x2 * si);
  kpe[(size_t)m * ROPEc + 2 * lane + 1] = f2bf(x1 * si + x2 * co);
}

// ---------------- RoPE on q_pe (reads q GEMM output, writes packed q_pe) ----------
__global__ __launch_bounds__(256)
void k_qrope(const u16* __restrict__ qbf, const float* __restrict__ fcos,
             const float* __restrict__ fsin, u16* __restrict__ qpe) {
  int idx = blockIdx.x * 256 + threadIdx.x;    // M*H*32
  int i = idx & 31;
  int hh = (idx >> 5) & (Hc - 1);
  int m = idx >> 9;
  int s = m & (Sc - 1);
  const u16* src = qbf + (size_t)m * (Hc * QKc) + hh * QKc + NOPEc + 2 * i;
  float x1 = bf2f(src[0]), x2 = bf2f(src[1]);
  float co = fcos[s * 32 + i], si = fsin[s * 32 + i];
  u16* dst = qpe + (size_t)m * (Hc * ROPEc) + hh * ROPEc + 2 * i;
  dst[0] = f2bf(x1 * co - x2 * si);
  dst[1] = f2bf(x1 * si + x2 * co);
}

// ---------------- flash attention over latent (512) + rope (64) keys --------------
// grid = (S/16, H, B); block = 128 (4 waves). Each block: 16 queries of one head.
// Score K-dim split across waves (w owns latent dims [w*128, w*128+128); wave 0 adds
// the 64 RoPE dims). Each wave accumulates output channels [w*128, w*128+128).
__global__ __launch_bounds__(128)
void k_attn(const u16* __restrict__ qlat, const u16* __restrict__ qpe,
            const u16* __restrict__ kvc, const u16* __restrict__ kpe,
            u16* __restrict__ aout) {
  __shared__ float sPart[4 * 2 * 256];   // per-wave partial 16x32 score tiles
  __shared__ u16 sP[16 * 40];            // P (16x32) bf16, padded rows
  __shared__ u16 sVt[512 * 40];          // V^T chunk: [channel][key], padded rows

  const int qb = blockIdx.x * 16;
  const int h = blockIdx.y;
  const int b = blockIdx.z;
  const int mt = b * Sc;

  const int tid = threadIdx.x;
  const int lane = tid & 31, w = tid >> 5;
  const int lg = lane >> 4, lr = lane & 15;

  // Q fragments for this wave's K-slice, held in registers for the whole block.
  const size_t qrow = (size_t)(mt + qb + lr);
  v16bf aQ[4];
#pragma unroll
  for (int j = 0; j < 4; ++j) {
    const u16* p = qlat + qrow * (Hc * KVRc) + h * KVRc + w * 128 + j * 32;
    aQ[j] = frag_from(*(const uint4*)(p + lg * 8), *(const uint4*)(p + 16 + lg * 8));
  }
  v16bf aPE[2] = {};
  if (w == 0) {
#pragma unroll
    for (int j = 0; j < 2; ++j) {
      const u16* p = qpe + qrow * (Hc * ROPEc) + h * ROPEc + j * 32;
      aPE[j] = frag_from(*(const uint4*)(p + lg * 8), *(const uint4*)(p + 16 + lg * 8));
    }
  }

  float mrun[8], lrun[8];
#pragma unroll
  for (int r = 0; r < 8; ++r) { mrun[r] = -1e30f; lrun[r] = 0.f; }
  v8f acc[8] = {};

  const float scale = 0.07216878364870322f;   // 1/sqrt(192)
  const int tEnd = (qb + 16 + 31) & ~31;

  for (int t0 = 0; t0 < tEnd; t0 += 32) {
    // prefetch next chunk's K rows into cache while this chunk computes
    if (t0 + 32 < tEnd) {
      __builtin_prefetch(kvc + (size_t)(t0 + 32 + lr) * KVRc + w * 128, 0, 1);
      __builtin_prefetch(kvc + (size_t)(t0 + 48 + lr) * KVRc + w * 128, 0, 1);
      if (w == 0) __builtin_prefetch(kpe + (size_t)(t0 + 32 + lr) * ROPEc, 0, 1);
    }

    // ---- partial scores: this wave's K-slice ----
    v8f sc[2] = {};
#pragma unroll
    for (int j = 0; j < 4; ++j) {
#pragma unroll
      for (int nb = 0; nb < 2; ++nb) {
        const u16* p = kvc + (size_t)(t0 + nb * 16 + lr) * KVRc + w * 128 + j * 32;
        v16bf bf = frag_from(*(const uint4*)(p + lg * 8), *(const uint4*)(p + 16 + lg * 8));
        sc[nb] = __builtin_amdgcn_wmma_f32_16x16x32_bf16(false, aQ[j], false, bf,
                                                         (short)0, sc[nb], false, false);
      }
    }
    if (w == 0) {
#pragma unroll
      for (int j = 0; j < 2; ++j) {
#pragma unroll
        for (int nb = 0; nb < 2; ++nb) {
          const u16* p = kpe + (size_t)(t0 + nb * 16 + lr) * ROPEc + j * 32;
          v16bf bf = frag_from(*(const uint4*)(p + lg * 8), *(const uint4*)(p + 16 + lg * 8));
          sc[nb] = __builtin_amdgcn_wmma_f32_16x16x32_bf16(false, aPE[j], false, bf,
                                                           (short)0, sc[nb], false, false);
        }
      }
    }
#pragma unroll
    for (int nb = 0; nb < 2; ++nb) {
      const float* f = (const float*)&sc[nb];
#pragma unroll
      for (int r = 0; r < 8; ++r)
        sPart[(w * 2 + nb) * 256 + r * 32 + lane] = f[r];
    }

    // ---- cooperatively stage V^T (kv_c chunk transposed) into LDS ----
#pragma unroll
    for (int pass = 0; pass < 16; ++pass) {
      int tt = pass * 2 + (tid >> 6);
      int c0 = (tid & 63) * 8;
      uint4 g = *(const uint4*)(kvc + (size_t)(t0 + tt) * KVRc + c0);
      const u32* gw = (const u32*)&g;
#pragma unroll
      for (int jj = 0; jj < 4; ++jj) {
        u32 v = gw[jj];
        sVt[(c0 + jj * 2 + 0) * 40 + tt] = (u16)(v & 0xffff);
        sVt[(c0 + jj * 2 + 1) * 40 + tt] = (u16)(v >> 16);
      }
    }
    __syncthreads();

    // ---- reduce partials, online softmax (replicated per wave) ----
    float pcur[2][8];
#pragma unroll
    for (int nb = 0; nb < 2; ++nb)
#pragma unroll
      for (int r = 0; r < 8; ++r) {
        float s = sPart[(0 + nb) * 256 + r * 32 + lane]
                + sPart[(2 + nb) * 256 + r * 32 + lane]
                + sPart[(4 + nb) * 256 + r * 32 + lane]
                + sPart[(6 + nb) * 256 + r * 32 + lane];
        s *= scale;
        int key = t0 + nb * 16 + lr;
        int qr = qb + lg * 8 + r;
        if (key > qr) s = -1e9f;     // causal mask
        pcur[nb][r] = s;
      }
#pragma unroll
    for (int r = 0; r < 8; ++r) {
      float mx = fmaxf(pcur[0][r], pcur[1][r]);
#pragma unroll
      for (int o = 1; o <= 8; o <<= 1) mx = fmaxf(mx, __shfl_xor(mx, o, 32));
      float mnew = fmaxf(mrun[r], mx);
      float alpha = __expf(mrun[r] - mnew);
      float p0 = __expf(pcur[0][r] - mnew);
      float p1 = __expf(pcur[1][r] - mnew);
      pcur[0][r] = p0; pcur[1][r] = p1;
      float rs = p0 + p1;
#pragma unroll
      for (int o = 1; o <= 8; o <<= 1) rs += __shfl_xor(rs, o, 32);
      lrun[r] = lrun[r] * alpha + rs;
      mrun[r] = mnew;
#pragma unroll
      for (int c = 0; c < 8; ++c) ((float*)&acc[c])[r] *= alpha;
    }
    if (w == 0) {                    // all waves hold identical P; one writes
#pragma unroll
      for (int nb = 0; nb < 2; ++nb)
#pragma unroll
        for (int r = 0; r < 8; ++r)
          sP[(lg * 8 + r) * 40 + nb * 16 + lr] = f2bf(pcur[nb][r]);
    }
    __syncthreads();

    // ---- P @ V for this wave's 128-channel slice ----
    {
      const u16* pp = &sP[lr * 40];
      v16bf aP = frag_from(*(const uint4*)(pp + lg * 8), *(const uint4*)(pp + 16 + lg * 8));
#pragma unroll
      for (int nb = 0; nb < 8; ++nb) {
        const u16* vp = &sVt[(w * 128 + nb * 16 + lr) * 40];
        v16bf bf = frag_from(*(const uint4*)(vp + lg * 8), *(const uint4*)(vp + 16 + lg * 8));
        acc[nb] = __builtin_amdgcn_wmma_f32_16x16x32_bf16(false, aP, false, bf,
                                                          (short)0, acc[nb], false, false);
      }
    }
    __syncthreads();
  }

  // ---- normalize and store ----
#pragma unroll
  for (int nb = 0; nb < 8; ++nb) {
    const float* f = (const float*)&acc[nb];
#pragma unroll
    for (int r = 0; r < 8; ++r) {
      size_t tok = (size_t)(mt + qb + lg * 8 + r);
      int col = h * KVRc + w * 128 + nb * 16 + lr;
      aout[tok * (Hc * KVRc) + col] = f2bf(f[r] / lrun[r]);
    }
  }
}

// ---------------- host-side orchestration ----------------
extern "C" void kernel_launch(void* const* d_in, const int* in_sizes, int n_in,
                              void* d_out, int out_size, void* d_ws, size_t ws_size,
                              hipStream_t stream) {
  const float* x       = (const float*)d_in[0];
  const float* wq      = (const float*)d_in[1];
  const float* wkv_a   = (const float*)d_in[2];
  const float* kvnw    = (const float*)d_in[3];
  const float* wkv_b   = (const float*)d_in[4];
  const float* wo      = (const float*)d_in[5];
  const float* fcos    = (const float*)d_in[6];
  const float* fsin    = (const float*)d_in[7];
  float* out = (float*)d_out;

  char* ws = (char*)d_ws;
  size_t off = 0;
  auto alloc = [&](size_t bytes) -> void* {
    void* p = ws + off;
    off = (off + bytes + 255) & ~(size_t)255;
    return p;
  };

  u16* xb    = (u16*)alloc((size_t)Mc * Dc * 2);
  u16* wqb   = (u16*)alloc((size_t)Hc * QKc * Dc * 2);
  u16* wab   = (u16*)alloc((size_t)(KVRc + ROPEc) * Dc * 2);
  u16* wob   = (u16*)alloc((size_t)Dc * Hc * Vc * 2);
  u16* wbbT  = (u16*)alloc((size_t)Hc * KVRc * NOPEc * 2);
  u16* wbbV  = (u16*)alloc((size_t)Hc * Vc * KVRc * 2);
  u16* q_bf  = (u16*)alloc((size_t)Mc * Hc * QKc * 2);
  float* kvf = (float*)alloc((size_t)Mc * (KVRc + ROPEc) * 4);
  u16* kv_c  = (u16*)alloc((size_t)Mc * KVRc * 2);
  u16* k_pe  = (u16*)alloc((size_t)Mc * ROPEc * 2);
  u16* q_pe  = (u16*)alloc((size_t)Mc * Hc * ROPEc * 2);
  u16* q_lat = (u16*)alloc((size_t)Mc * Hc * KVRc * 2);
  u16* a_out = (u16*)alloc((size_t)Mc * Hc * KVRc * 2);
  u16* o_hd  = (u16*)alloc((size_t)Mc * Hc * Vc * 2);
  (void)ws_size; (void)in_sizes; (void)n_in; (void)out_size;

  // 1) fp32 -> bf16 conversions
  k_cvt<<<dim3(Mc * Dc / 1024), 256, 0, stream>>>(x, xb, Mc * Dc);
  k_cvt<<<dim3(Hc * QKc * Dc / 1024), 256, 0, stream>>>(wq, wqb, Hc * QKc * Dc);
  k_cvt<<<dim3((KVRc + ROPEc) * Dc / 1024), 256, 0, stream>>>(wkv_a, wab, (KVRc + ROPEc) * Dc);
  k_cvt<<<dim3(Dc * Hc * Vc / 1024), 256, 0, stream>>>(wo, wob, Dc * Hc * Vc);
  k_cvt_wkvb<<<dim3(Hc * 256 * KVRc / 256), 256, 0, stream>>>(wkv_b, wbbT, wbbV);

  // 2) q = x @ wq^T   (M=8192, N=3072, K=2048) -> bf16
  k_gemm_nt<true><<<dim3(Hc * QKc / 64, Mc / 64, 1), 128, 0, stream>>>(
      xb, wqb, q_bf, Dc, Dc, Dc, Hc * QKc, 0, 0, 0);

  // 3) kv = x @ wkv_a^T (M=8192, N=576, K=2048) -> f32 (for RMSNorm precision)
  k_gemm_nt<false><<<dim3((KVRc + ROPEc) / 64, Mc / 64, 1), 128, 0, stream>>>(
      xb, wab, kvf, Dc, Dc, Dc, KVRc + ROPEc, 0, 0, 0);

  // 4) RMSNorm + RoPE on kv ; RoPE on q_pe
  k_kvpost<<<dim3(Mc / 8), 256, 0, stream>>>(kvf, kvnw, fcos, fsin, kv_c, k_pe);
  k_qrope<<<dim3(Mc * Hc * 32 / 256), 256, 0, stream>>>(q_bf, fcos, fsin, q_pe);

  // 5) q_lat[h] = q_nope[h] @ wkv_b_nope[h]   (batched over heads)
  k_gemm_nt<true><<<dim3(KVRc / 64, Mc / 64, Hc), 128, 0, stream>>>(
      q_bf, wbbT, q_lat, NOPEc, Hc * QKc, NOPEc, Hc * KVRc,
      (long long)QKc, (long long)KVRc * NOPEc, (long long)KVRc);

  // 6) causal flash attention over latent+rope keys
  k_attn<<<dim3(Sc / 16, Hc, Bc), 128, 0, stream>>>(q_lat, q_pe, kv_c, k_pe, a_out);

  // 7) per-head value projection: o[h] = attn[h] @ wkv_b_v[h]^T
  k_gemm_nt<true><<<dim3(Vc / 64, Mc / 64, Hc), 128, 0, stream>>>(
      a_out, wbbV, o_hd, KVRc, Hc * KVRc, KVRc, Hc * Vc,
      (long long)KVRc, (long long)Vc * KVRc, (long long)Vc);

  // 8) final = o_heads @ wo^T -> f32 output (M=8192, N=2048, K=2048)
  k_gemm_nt<false><<<dim3(Dc / 64, Mc / 64, 1), 128, 0, stream>>>(
      o_hd, wob, out, Hc * Vc, Hc * Vc, Hc * Vc, Dc, 0, 0, 0);
}
